// SlidingWindowAttention_31662498906977
// MI455X (gfx1250) — compile-verified, matrix operand order checked by reading, not run on hardware
//
#include <hip/hip_runtime.h>
#include <hip/hip_bf16.h>

typedef __attribute__((ext_vector_type(16))) __bf16 bf16x16;
typedef __attribute__((ext_vector_type(8)))  float  f32x8;

union FragU { bf16x16 v; uint4 q[2]; };

__device__ __forceinline__ f32x8 zero8() {
    f32x8 z = {0.f, 0.f, 0.f, 0.f, 0.f, 0.f, 0.f, 0.f};
    return z;
}

// Load a 16x32 bf16 WMMA operand fragment from LDS (row-major, `stride` bf16 per row).
// Per ISA 7.12.2: lanes 0-15 hold row r=lane, k-phases {0..7, 16..23};
// lanes 16-31 hold row r=lane-16, k-phases {8..15, 24..31}.
__device__ __forceinline__ bf16x16 frag_ld(const __bf16* base, int stride, int lane) {
    const int r  = lane & 15;
    const int ph = (lane & 16) ? 8 : 0;
    FragU u;
    u.q[0] = *(const uint4*)(base + r * stride + ph);
    u.q[1] = *(const uint4*)(base + r * stride + ph + 16);
    return u.v;
}

__device__ __forceinline__ f32x8 wmma_bf16(bf16x16 a, bf16x16 b, f32x8 c) {
    return __builtin_amdgcn_wmma_f32_16x16x32_bf16(false, a, false, b, (short)0, c,
                                                   false, false);
}

// Async 16B global->LDS copy (ASYNCcnt-tracked; no VGPR data round-trip).
// LDS dest operand: hardware adds LDS_BASE, and flat-aperture rules make the low
// 32 bits of a generic shared pointer the wave-relative LDS byte address.
__device__ __forceinline__ void async_cp_b128(const void* gsrc, void* ldst) {
    unsigned loff = (unsigned)(unsigned long long)ldst;
    asm volatile("global_load_async_to_lds_b128 %0, %1, off"
                 :: "v"(loff), "v"(gsrc) : "memory");
}

__device__ __forceinline__ void wait_async0() {
#if __has_builtin(__builtin_amdgcn_s_wait_asynccnt)
    __builtin_amdgcn_s_wait_asynccnt(0);
#else
    asm volatile("s_wait_asynccnt 0" ::: "memory");
#endif
}

// -------- GEMM: C[M,N] = A[M,K] (f32, row-major) * B[K,N] (f32, row-major) --------
// Block tile 128(M) x 256(N), BK=32, 256 threads (8 waves, 2x4), wave tile 64x64:
// 16 WMMAs per 8 fragment loads per K-step.
template <typename CT>
__global__ __launch_bounds__(256) void gemm_bf16_wmma(const float* __restrict__ A,
                                                      const float* __restrict__ B,
                                                      CT* __restrict__ C,
                                                      int M, int N, int K) {
    constexpr int KP = 40;  // padded K-stride (80B rows: conflict-free frag gathers)
    __shared__ __align__(16) __bf16 As[128 * KP];
    __shared__ __align__(16) __bf16 Bs[256 * KP];  // stored transposed: Bs[n][k]

    const int t    = threadIdx.x;
    const int lane = t & 31;
    const int wave = t >> 5;
    const int mw   = (wave & 1) * 64;
    const int nw   = (wave >> 1) * 64;
    const int m0   = blockIdx.y * 128;
    const int n0   = blockIdx.x * 256;

    f32x8 acc[4][4];
#pragma unroll
    for (int i = 0; i < 4; ++i)
#pragma unroll
        for (int j = 0; j < 4; ++j) acc[i][j] = zero8();

    for (int kb = 0; kb < K; kb += 32) {
        __syncthreads();
        // Stage A: 128 rows x 32 k, float4 per slot (1024 slots / 256 threads).
#pragma unroll
        for (int it = 0; it < 4; ++it) {
            int slot = t + it * 256;
            int row = slot >> 3, c4 = slot & 7;
            float4 f = *(const float4*)(A + (size_t)(m0 + row) * K + kb + c4 * 4);
            __bf16* dst = &As[row * KP + c4 * 4];
            dst[0] = (__bf16)f.x; dst[1] = (__bf16)f.y;
            dst[2] = (__bf16)f.z; dst[3] = (__bf16)f.w;
        }
        // Stage B transposed: 32 k-rows x 256 n (2048 float4 slots).
#pragma unroll
        for (int it = 0; it < 8; ++it) {
            int slot = t + it * 256;
            int kr = slot >> 6, n4 = slot & 63;
            float4 f = *(const float4*)(B + (size_t)(kb + kr) * N + n0 + n4 * 4);
            Bs[(n4 * 4 + 0) * KP + kr] = (__bf16)f.x;
            Bs[(n4 * 4 + 1) * KP + kr] = (__bf16)f.y;
            Bs[(n4 * 4 + 2) * KP + kr] = (__bf16)f.z;
            Bs[(n4 * 4 + 3) * KP + kr] = (__bf16)f.w;
        }
        __syncthreads();

        bf16x16 af[4];
#pragma unroll
        for (int i = 0; i < 4; ++i) af[i] = frag_ld(&As[(mw + i * 16) * KP], KP, lane);
#pragma unroll
        for (int j = 0; j < 4; ++j) {
            bf16x16 bfr = frag_ld(&Bs[(nw + j * 16) * KP], KP, lane);
#pragma unroll
            for (int i = 0; i < 4; ++i) acc[i][j] = wmma_bf16(af[i], bfr, acc[i][j]);
        }
    }

    // C/D layout: lanes 0-15 -> M = r, N = lane; lanes 16-31 -> M = r+8, N = lane-16.
    const int rbase = m0 + mw + ((lane & 16) ? 8 : 0);
    const int cbase = n0 + nw + (lane & 15);
#pragma unroll
    for (int i = 0; i < 4; ++i)
#pragma unroll
        for (int j = 0; j < 4; ++j)
#pragma unroll
            for (int r = 0; r < 8; ++r)
                C[(size_t)(rbase + i * 16 + r) * N + cbase + j * 16] = (CT)acc[i][j][r];
}

// -------- Sliding-window attention (W=256), flash-style with WMMA --------
// qkv: bf16 [B*S, 3072] (q | k | v, each head-major h*64+d). attn: f32 [B*S, 1024].
// Grid: (S/64, H, B); 128 threads = 4 waves; wave handles 16 queries.
__global__ __launch_bounds__(128) void attn_swa(const __bf16* __restrict__ qkv,
                                                float* __restrict__ attnOut) {
    constexpr int S = 2048, DM = 1024, WIN = 256;
    constexpr float SCALE = 0.125f;  // 1/sqrt(64)

    __shared__ __align__(16) __bf16 Qs[64 * 72];
    __shared__ __align__(16) __bf16 Ks[32 * 72];   // K chunk, row-major [key][d]
    __shared__ __align__(16) __bf16 Vt[64 * 40];   // V chunk transposed [d][key]
    __shared__ __align__(16) __bf16 Pw[4][16 * 40];

    const int t    = threadIdx.x;
    const int lane = t & 31;
    const int w    = t >> 5;
    const int q0   = blockIdx.x * 64;
    const int h    = blockIdx.y;
    const int b    = blockIdx.z;
    const size_t rowbase = (size_t)b * S;

    // Stage Q block (64 x 64 bf16) via async global->LDS b128 copies.
#pragma unroll
    for (int it = 0; it < 4; ++it) {
        int slot = t + it * 128;
        int row = slot >> 3, c = slot & 7;
        async_cp_b128(qkv + (rowbase + q0 + row) * 3072 + h * 64 + c * 8,
                      &Qs[row * 72 + c * 8]);
    }
    wait_async0();
    __syncthreads();

    const int qw = q0 + w * 16;
    const int qoff = (lane & 16) ? 8 : 0;
    bf16x16 qa0 = frag_ld(&Qs[(w * 16) * 72], 72, lane);
    bf16x16 qa1 = frag_ld(&Qs[(w * 16) * 72 + 32], 72, lane);

    f32x8 O[4];
#pragma unroll
    for (int j = 0; j < 4; ++j) O[j] = zero8();
    float Mr[8], Lr[8];
#pragma unroll
    for (int r = 0; r < 8; ++r) { Mr[r] = -1e30f; Lr[r] = 0.f; }

    const int klo = (q0 >= WIN) ? (q0 - WIN) : 0;

    for (int kbase = klo; kbase < q0 + 64; kbase += 32) {
        __syncthreads();
        // Stage K chunk (32 keys x 64 d) via async copies.
#pragma unroll
        for (int it = 0; it < 2; ++it) {
            int slot = t + it * 128;
            int row = slot >> 3, c = slot & 7;
            async_cp_b128(qkv + (rowbase + kbase + row) * 3072 + DM + h * 64 + c * 8,
                          &Ks[row * 72 + c * 8]);
        }
        // Stage V chunk transposed: Vt[d][key] (transpose during copy -> scalar path).
#pragma unroll
        for (int i2 = 0; i2 < 16; ++i2) {
            int idx = t + i2 * 128;
            int key = idx >> 6, dd = idx & 63;
            Vt[dd * 40 + key] = qkv[(rowbase + kbase + key) * 3072 + 2 * DM + h * 64 + dd];
        }
        wait_async0();
        __syncthreads();

        // Wave-uniform skip if chunk is fully outside this wave's window.
        if (kbase > qw + 15 || kbase + 31 < qw - (WIN - 1)) continue;

        // Scores for both 16-key sub-tiles: S = Q * K^T (K=64 -> two WMMAs each).
        f32x8 sc[2];
#pragma unroll
        for (int st = 0; st < 2; ++st) {
            bf16x16 kb0 = frag_ld(&Ks[(st * 16) * 72], 72, lane);
            bf16x16 kb1 = frag_ld(&Ks[(st * 16) * 72 + 32], 72, lane);
            f32x8 s = zero8();
            s = wmma_bf16(qa0, kb0, s);
            s = wmma_bf16(qa1, kb1, s);
            sc[st] = s;
        }

        // Mask: allowed iff 0 <= qi - kcol < WIN  (single unsigned range test).
        float sv[2][8];
        bool  ok[2][8];
#pragma unroll
        for (int st = 0; st < 2; ++st) {
            const int kcol = kbase + st * 16 + (lane & 15);
            const int d0   = qw + qoff - kcol;
#pragma unroll
            for (int r = 0; r < 8; ++r) {
                ok[st][r] = (unsigned)(d0 + r) < (unsigned)WIN;
                sv[st][r] = ok[st][r] ? sc[st][r] * SCALE : -1e30f;
            }
        }

        // Online softmax: per-query stats are uniform across each 16-lane half,
        // and the C-layout (vgpr=r, half=lane>=16) maps to the same query in both
        // the score fragment and the O fragment, so fsc applies directly to O.
        float fsc[8];
#pragma unroll
        for (int r = 0; r < 8; ++r) {
            float v = fmaxf(sv[0][r], sv[1][r]);
#pragma unroll
            for (int mk = 1; mk < 16; mk <<= 1) v = fmaxf(v, __shfl_xor(v, mk, 32));
            float nM  = fmaxf(Mr[r], v);
            float f   = __expf(Mr[r] - nM);
            float pe0 = ok[0][r] ? __expf(sv[0][r] - nM) : 0.f;
            float pe1 = ok[1][r] ? __expf(sv[1][r] - nM) : 0.f;
            float ps  = pe0 + pe1;
#pragma unroll
            for (int mk = 1; mk < 16; mk <<= 1) ps += __shfl_xor(ps, mk, 32);
            Lr[r] = Lr[r] * f + ps;
            Mr[r] = nM;
            fsc[r] = f;
            Pw[w][(qoff + r) * 40 + (lane & 15)]      = (__bf16)pe0;
            Pw[w][(qoff + r) * 40 + 16 + (lane & 15)] = (__bf16)pe1;
        }
#pragma unroll
        for (int j = 0; j < 4; ++j)
#pragma unroll
            for (int r = 0; r < 8; ++r) O[j][r] *= fsc[r];

        // O += P * V  (A = P 16x32, B = V 32x16 per d-tile; 4 WMMAs).
        bf16x16 pf = frag_ld(&Pw[w][0], 40, lane);
#pragma unroll
        for (int j = 0; j < 4; ++j) {
            bf16x16 vb = frag_ld(&Vt[(j * 16) * 40], 40, lane);
            O[j] = wmma_bf16(pf, vb, O[j]);
        }
    }

#pragma unroll
    for (int r = 0; r < 8; ++r) Lr[r] = 1.f / Lr[r];
#pragma unroll
    for (int j = 0; j < 4; ++j)
#pragma unroll
        for (int r = 0; r < 8; ++r) {
            int row = qw + qoff + r;
            int col = h * 64 + j * 16 + (lane & 15);
            attnOut[(rowbase + row) * DM + col] = O[j][r] * Lr[r];
        }
}

extern "C" void kernel_launch(void* const* d_in, const int* in_sizes, int n_in,
                              void* d_out, int out_size, void* d_ws, size_t ws_size,
                              hipStream_t stream) {
    const float* x    = (const float*)d_in[0];
    const float* Wqkv = (const float*)d_in[1];
    const float* Wout = (const float*)d_in[2];
    float* out = (float*)d_out;

    constexpr int B = 2, S = 2048, DM = 1024;
    constexpr size_t qkvBytes = (size_t)B * S * 3 * DM * sizeof(__bf16);  // ~25 MB
    __bf16* qkv = (__bf16*)d_ws;
    float* attn = (float*)((char*)d_ws + ((qkvBytes + 255) & ~(size_t)255));  // ~17 MB

    // 1) QKV projection: [4096,1024] x [1024,3072] -> bf16 qkv
    gemm_bf16_wmma<__bf16><<<dim3(3 * DM / 256, B * S / 128), 256, 0, stream>>>(
        x, Wqkv, qkv, B * S, 3 * DM, DM);
    // 2) Sliding-window attention -> f32 attn [4096,1024] (head-major cols)
    attn_swa<<<dim3(S / 64, 16, B), 128, 0, stream>>>(qkv, attn);
    // 3) Output projection: [4096,1024] x [1024,1024] -> f32 out
    gemm_bf16_wmma<float><<<dim3(DM / 256, B * S / 128), 256, 0, stream>>>(
        attn, Wout, out, B * S, DM, DM);
}
